// SubmatrixLinear_41223096107797
// MI455X (gfx1250) — matrix operand order repaired and checked by reading, not compiled, not run on hardware
//
#include <hip/hip_runtime.h>

// ---------------------------------------------------------------------------
// SubmatrixLinear on MI455X (gfx1250, wave32, WMMA + async-to-LDS):
//   out = x@W^T + b  fused with  delta = (gates ⊙ (B x)) · A
// x and W are pre-converted to bf16 in workspace (one-time pass); the main
// GEMM streams 128x64 bf16 tiles to LDS with global_load_async_to_lds_b128 in
// GVS mode (per-thread VGPR offsets fixed, SGPR base advances 128B per block),
// double-buffered via ASYNCcnt, computing v_wmma_f32_16x16x32_bf16 (f32 acc).
// The rank K*R=64 delta GEMM is fused as exactly ONE extra pipeline block.
// Workspace: gates 256K + wqT 256K + x_bf 32M + w_bf 8M + gbx_bf 1M + a2_bf 256K
//            ≈ 42 MB (d_ws must be at least this large).
// ---------------------------------------------------------------------------

typedef __attribute__((ext_vector_type(16))) __bf16 v16bf;
typedef __attribute__((ext_vector_type(8)))  float  v8f;

#define BATCH 8192
#define DIN   2048
#define DOUT  2048
#define RK    64      // K*R = 8*8
#define NKEY  8
#define KD    32      // KEY_DIM

__device__ __forceinline__ unsigned short f2bf(float f) {
  union { __bf16 h; unsigned short u; } c;
  c.h = (__bf16)f;                                   // hw RNE convert
  return c.u;
}

// ---------------- tiny prep kernels ----------------------------------------

__global__ void k_transpose_wq(const float* __restrict__ wq, float* __restrict__ wqT) {
  int i = blockIdx.x * 256 + threadIdx.x;            // 32*2048 = 65536 exact
  int r = i >> 11, c = i & 2047;                     // wq is (32, 2048)
  wqT[c * KD + r] = wq[i];
}

// f32 -> bf16, 4 elements per thread (n multiple of 1024)
__global__ void k_cvt_bf16(const float* __restrict__ src, unsigned short* __restrict__ dst) {
  int i = (blockIdx.x * 256 + threadIdx.x) * 4;
  float4 v = *(const float4*)(src + i);
  uint2 p;
  p.x = (unsigned)f2bf(v.x) | ((unsigned)f2bf(v.y) << 16);
  p.y = (unsigned)f2bf(v.z) | ((unsigned)f2bf(v.w) << 16);
  *(uint2*)(dst + i) = p;
}

__global__ void k_build_a2(const float* __restrict__ A, unsigned short* __restrict__ a2) {
  int i = blockIdx.x * 256 + threadIdx.x;            // 2048*64 = 131072 exact
  int o = i >> 6, kr = i & 63;
  int k = kr >> 3, r = kr & 7;
  a2[i] = f2bf(A[(size_t)(k * DOUT + o) * 8 + r]);   // A2[o][k*8+r] = A[k][o][r]
}

// ---------------- gates: q = x Wq^T + bq ; sparsemax(q keys^T / sqrt(32)) --

__global__ void k_gates(const float* __restrict__ x, const float* __restrict__ wqT,
                        const float* __restrict__ bq, const float* __restrict__ keys,
                        float* __restrict__ gates) {
  int row  = (blockIdx.x * 256 + threadIdx.x) >> 5;  // one wave per batch row
  int lane = threadIdx.x & 31;
  const float* xr = x + (size_t)row * DIN;

  float q = 0.f;
  #pragma unroll 4
  for (int i = 0; i < DIN; ++i)
    q = fmaf(xr[i], wqT[i * KD + lane], q);          // xr[i] uniform -> scalar load
  q += bq[lane];

  // scores[k] = (1/sqrt(32)) * sum_l q_l * keys[k][l]   (butterfly all-reduce)
  float s[NKEY];
  #pragma unroll
  for (int k = 0; k < NKEY; ++k) {
    float t = q * keys[k * KD + lane];
    #pragma unroll
    for (int off = 16; off >= 1; off >>= 1) t += __shfl_xor(t, off, 32);
    s[k] = t * 0.17677669529663689f;
  }

  // sparsemax over 8 values (computed redundantly per lane)
  float z[NKEY];
  #pragma unroll
  for (int k = 0; k < NKEY; ++k) z[k] = s[k];
#define CSWAP(a, b) { if (z[a] < z[b]) { float t = z[a]; z[a] = z[b]; z[b] = t; } }
  CSWAP(0,1) CSWAP(2,3) CSWAP(4,5) CSWAP(6,7)
  CSWAP(0,2) CSWAP(1,3) CSWAP(4,6) CSWAP(5,7)
  CSWAP(1,2) CSWAP(5,6)
  CSWAP(0,4) CSWAP(1,5) CSWAP(2,6) CSWAP(3,7)
  CSWAP(2,4) CSWAP(3,5)
  CSWAP(1,2) CSWAP(3,4) CSWAP(5,6)
#undef CSWAP
  float c[NKEY];
  c[0] = z[0];
  #pragma unroll
  for (int j = 1; j < NKEY; ++j) c[j] = c[j - 1] + z[j];
  int ksel = 1; float tausum = c[0];
  #pragma unroll
  for (int j = 1; j < NKEY; ++j)
    if (1.0f + (float)(j + 1) * z[j] > c[j]) { ksel = j + 1; tausum = c[j]; }
  float tau = (tausum - 1.0f) / (float)ksel;

  if (lane == 0) {
    #pragma unroll
    for (int k = 0; k < NKEY; ++k)
      gates[(size_t)row * NKEY + k] = fmaxf(s[k] - tau, 0.0f);
  }
}

// -------- gbx[b][kr] = bf16( gates[b][k] * sum_i x[b][i] * B[k][r][i] ) ----

__global__ void k_gbx(const float* __restrict__ x, const float* __restrict__ B,
                      const float* __restrict__ gates, unsigned short* __restrict__ gbx) {
  int row  = (blockIdx.x * 256 + threadIdx.x) >> 5;  // one wave per batch row
  int lane = threadIdx.x & 31;
  const float* xr = x + (size_t)row * DIN;
  int o0 = 2 * lane, o1 = 2 * lane + 1;              // two kr-outputs per lane
  const float* b0 = B + (size_t)o0 * DIN;
  const float* b1 = B + (size_t)o1 * DIN;
  float a0 = 0.f, a1 = 0.f;
  #pragma unroll 4
  for (int i = 0; i < DIN; ++i) {
    float xv = xr[i];
    a0 = fmaf(xv, b0[i], a0);
    a1 = fmaf(xv, b1[i], a1);
  }
  float g = gates[(size_t)row * NKEY + (o0 >> 3)];   // o0,o1 share the same k
  gbx[(size_t)row * RK + o0] = f2bf(a0 * g);
  gbx[(size_t)row * RK + o1] = f2bf(a1 * g);
}

// ---------------- main fused WMMA GEMM with async-to-LDS -------------------

#define BM 128
#define BN 128
#define BK 64     // two WMMA K-steps per staged tile
#define LDSS 72   // shorts per LDS row (144B; 16B-aligned, conflict-skewed)

union FragBf { v16bf v; uint4 q[2]; };

// GVS-mode async copy: saddr(SGPR base) + vaddr(32-bit VGPR byte offset) -> LDS
__device__ __forceinline__ void async_copy16(unsigned lds_off, unsigned voff,
                                             const unsigned short* sbase) {
  asm volatile("global_load_async_to_lds_b128 %0, %1, %2"
               :: "v"(lds_off), "v"(voff), "s"(sbase) : "memory");
}
__device__ __forceinline__ void wait_async_le8() {
  asm volatile("s_wait_asynccnt 0x8" ::: "memory");
}
__device__ __forceinline__ void wait_async_0() {
  asm volatile("s_wait_asynccnt 0x0" ::: "memory");
}

__global__ void __launch_bounds__(256)
k_gemm(const unsigned short* __restrict__ xb, const unsigned short* __restrict__ wb,
       const float* __restrict__ bias, const unsigned short* __restrict__ gbx,
       const unsigned short* __restrict__ a2, float* __restrict__ out) {
  __shared__ __align__(16) unsigned short ldsA[2][BM * LDSS];  // 2 x 18KB
  __shared__ __align__(16) unsigned short ldsB[2][BN * LDSS];  // 2 x 18KB

  const int tid  = threadIdx.x;
  const int lane = tid & 31;
  const int wave = tid >> 5;
  const int wr = wave >> 2;           // 2 wave-rows  (64 rows each)
  const int wc = wave & 3;            // 4 wave-cols  (32 cols each)
  const int row0 = blockIdx.y * BM;   // batch rows
  const int col0 = blockIdx.x * BN;   // output cols

  // Per-thread copy geometry: 128x64 bf16 tile = 1024 16B-chunks; 4 per thread.
  const int kc = (tid & 7) * 8;       // bf16 column within 64-wide K-block
  unsigned lo[4], vm[4], vt[4];
  #pragma unroll
  for (int j = 0; j < 4; ++j) {
    int r = (tid >> 3) + 32 * j;      // rows 0..127
    lo[j] = (unsigned)(r * LDSS + kc) * 2u;   // LDS byte offset
    vm[j] = (unsigned)(r * DIN  + kc) * 2u;   // main source (ld = DIN)
    vt[j] = (unsigned)(r * RK   + kc) * 2u;   // tail source (ld = RK)
  }

  // LDS byte offsets of the double-buffers (low 32 bits of generic ptr)
  const unsigned aoff[2] = { (unsigned)(unsigned long long)&ldsA[0][0],
                             (unsigned)(unsigned long long)&ldsA[1][0] };
  const unsigned boff[2] = { (unsigned)(unsigned long long)&ldsB[0][0],
                             (unsigned)(unsigned long long)&ldsB[1][0] };

  // Uniform SGPR bases; advance by BK elements (128 bytes) per issued block.
  const unsigned short* baseA = xb  + (size_t)row0 * DIN;
  const unsigned short* baseB = wb  + (size_t)col0 * DIN;
  const unsigned short* tailA = gbx + (size_t)row0 * RK;
  const unsigned short* tailB = a2  + (size_t)col0 * RK;

  v8f acc[4][2];
  #pragma unroll
  for (int mi = 0; mi < 4; ++mi)
    #pragma unroll
    for (int ni = 0; ni < 2; ++ni)
      acc[mi][ni] = (v8f){0.f,0.f,0.f,0.f,0.f,0.f,0.f,0.f};

  const int rsub  = lane & 15;
  const int khalf = lane >> 4;

  auto issue_main = [&](int buf) {                   // 8 async instr / wave
    #pragma unroll
    for (int j = 0; j < 4; ++j) {
      async_copy16(aoff[buf] + lo[j], vm[j], baseA);
      async_copy16(boff[buf] + lo[j], vm[j], baseB);
    }
    baseA += BK; baseB += BK;
  };
  auto issue_tail = [&](int buf) {                   // 8 async instr / wave
    #pragma unroll
    for (int j = 0; j < 4; ++j) {
      async_copy16(aoff[buf] + lo[j], vt[j], tailA);
      async_copy16(boff[buf] + lo[j], vt[j], tailB);
    }
  };

  auto compute = [&](int cur) {
    const unsigned short* pa = &ldsA[cur][0];
    const unsigned short* pb = &ldsB[cur][0];
    #pragma unroll
    for (int s = 0; s < 2; ++s) {                    // two 32-wide K-steps
      const int k0 = s * 32;
      // A frag: lane(row=rsub, khalf) holds K {khalf*8..+7} and {16+khalf*8..+7}
      FragBf fa[4], fb[2];
      #pragma unroll
      for (int mi = 0; mi < 4; ++mi) {
        const unsigned short* p = pa + (wr * 64 + mi * 16 + rsub) * LDSS + k0;
        fa[mi].q[0] = *(const uint4*)(p + khalf * 8);
        fa[mi].q[1] = *(const uint4*)(p + 16 + khalf * 8);
      }
      // B frag: lane(col=rsub, khalf) holds K {khalf*8..+7} and {16+khalf*8..+7}
      #pragma unroll
      for (int ni = 0; ni < 2; ++ni) {
        const unsigned short* p = pb + (wc * 32 + ni * 16 + rsub) * LDSS + k0;
        fb[ni].q[0] = *(const uint4*)(p + khalf * 8);
        fb[ni].q[1] = *(const uint4*)(p + 16 + khalf * 8);
      }
      #pragma unroll
      for (int mi = 0; mi < 4; ++mi)
        #pragma unroll
        for (int ni = 0; ni < 2; ++ni)
          acc[mi][ni] = __builtin_amdgcn_wmma_f32_16x16x32_bf16(
              false, fa[mi].v, false, fb[ni].v, (short)0, acc[mi][ni], false, false);
    }
  };

  // Pipeline: 32 main blocks (x/W, K=2048) + 1 tail block (gBx/A2, K=64).
  // Block j lives in buffer j&1; branch-free steady-state loop.
  issue_main(0);
  for (int kb = 0; kb < 31; ++kb) {
    const int cur = kb & 1;
    issue_main(cur ^ 1);
    wait_async_le8();                                // current group landed
    __syncthreads();
    compute(cur);
    __syncthreads();                                 // done reading before reuse
  }
  issue_tail(0);                                     // tail block 32 -> buf0
  wait_async_le8();
  __syncthreads();
  compute(1);                                        // main block 31
  __syncthreads();
  wait_async_0();
  __syncthreads();
  compute(0);                                        // fused low-rank tail

  // epilogue: +bias, store f32.  D layout: VGPR e, lanes<16 -> M=e, lanes>=16 -> M=8+e
  #pragma unroll
  for (int mi = 0; mi < 4; ++mi) {
    #pragma unroll
    for (int ni = 0; ni < 2; ++ni) {
      int col = col0 + wc * 32 + ni * 16 + rsub;
      float bv = bias[col];
      v8f a = acc[mi][ni];
      #pragma unroll
      for (int e = 0; e < 8; ++e) {
        int rowg = row0 + wr * 64 + mi * 16 + khalf * 8 + e;
        out[(size_t)rowg * DOUT + col] = a[e] + bv;
      }
    }
  }
}

// ---------------- launcher -------------------------------------------------

extern "C" void kernel_launch(void* const* d_in, const int* in_sizes, int n_in,
                              void* d_out, int out_size, void* d_ws, size_t ws_size,
                              hipStream_t stream) {
  const float* x    = (const float*)d_in[0];
  const float* Wb   = (const float*)d_in[1];
  const float* bb   = (const float*)d_in[2];
  const float* Wq   = (const float*)d_in[3];
  const float* bq   = (const float*)d_in[4];
  const float* keys = (const float*)d_in[5];
  const float* A    = (const float*)d_in[6];
  const float* B    = (const float*)d_in[7];
  float* out = (float*)d_out;

  char* ws = (char*)d_ws;
  size_t off = 0;
  float* gates = (float*)(ws + off);          off += (size_t)BATCH * NKEY * 4;      // 256 KB
  float* wqT   = (float*)(ws + off);          off += (size_t)DIN * KD * 4;          // 256 KB
  unsigned short* x_bf  = (unsigned short*)(ws + off); off += (size_t)BATCH * DIN * 2;  // 32 MB
  unsigned short* w_bf  = (unsigned short*)(ws + off); off += (size_t)DOUT * DIN * 2;   //  8 MB
  unsigned short* gbx   = (unsigned short*)(ws + off); off += (size_t)BATCH * RK * 2;   //  1 MB
  unsigned short* a2    = (unsigned short*)(ws + off);                                  // 256 KB

  k_transpose_wq<<<256, 256, 0, stream>>>(Wq, wqT);
  k_cvt_bf16    <<<(BATCH * DIN) / 1024, 256, 0, stream>>>(x, x_bf);
  k_cvt_bf16    <<<(DOUT * DIN) / 1024, 256, 0, stream>>>(Wb, w_bf);
  k_build_a2    <<<512, 256, 0, stream>>>(A, a2);
  k_gates       <<<1024, 256, 0, stream>>>(x, wqT, bq, keys, gates);
  k_gbx         <<<1024, 256, 0, stream>>>(x, B, gates, gbx);

  dim3 grid(DOUT / BN, BATCH / BM);           // 16 x 64 workgroups
  k_gemm<<<grid, 256, 0, stream>>>(x_bf, w_bf, bb, gbx, a2, out);
}
